// GCN_28595892256902
// MI455X (gfx1250) — matrix operand order repaired and checked by reading, not compiled, not run on hardware
//
#include <hip/hip_runtime.h>
#include <hip/hip_bf16.h>

#define GN 50000
#define GE 800000
#define GH 64
#define GOUT 32
#define GG 512
#define GEPS 1e-5f

typedef __attribute__((ext_vector_type(16))) _Float16 v16h;
typedef __attribute__((ext_vector_type(8)))  float    v8f;

// ---------------------------------------------------------------------------
// zero a float buffer
__global__ void zero_kernel(float* __restrict__ p, long long n) {
  long long t = (long long)blockIdx.x * blockDim.x + threadIdx.x;
  if (t < n) p[t] = 0.0f;
}

// ---------------------------------------------------------------------------
// degree of (A + I): atomic count of in-edges (self-loop +1 added in dinv)
__global__ void degree_kernel(const int* __restrict__ dst, float* __restrict__ deg, int e_cnt) {
  int e = blockIdx.x * blockDim.x + threadIdx.x;
  if (e < e_cnt) atomicAdd(&deg[dst[e]], 1.0f);
}

__global__ void dinv_kernel(float* __restrict__ deg, int n_cnt) {
  int n = blockIdx.x * blockDim.x + threadIdx.x;
  if (n < n_cnt) deg[n] = rsqrtf(deg[n] + 1.0f);
}

// ---------------------------------------------------------------------------
// h = x @ W   (N x 64) @ (64 x 64), f16 WMMA with f32 accumulate.
// Block = 256 threads = 8 waves; block covers 64 rows x 64 cols.
// Tile grid 4x4; wave w owns col tile (w&3) and row tiles {(w>>2), (w>>2)+2}.
// W staged in LDS pre-swizzled into B-fragment order (contiguous 32B/lane);
// X staged with float4 global loads. Store epilogue: block-uniform bounds
// test so 781/782 blocks take the straight-line 8-store path.
__global__ void gemm_xw_kernel(const float* __restrict__ X, const float* __restrict__ W,
                               float* __restrict__ Hout) {
  __shared__ _Float16 sA[64 * 64];   // x tile, row-major [m][k]
  __shared__ _Float16 sB[64 * 64];   // W swizzled: [(k>>4)][n][k&15]
  const int tid = threadIdx.x;       // 0..255
  const int m0  = blockIdx.x * 64;

  // stage X tile: 64 rows x 64 cols as float4 chunks (1024 chunks / 256 thr)
  for (int i = tid; i < 64 * 16; i += 256) {
    int r = i >> 4, c4 = (i & 15) * 4;
    int row = m0 + r; row = (row < GN) ? row : (GN - 1);   // clamp OOB rows
    float4 v = *(const float4*)(X + (size_t)row * GH + c4);
    _Float16* dp = sA + r * 64 + c4;
    dp[0] = (_Float16)v.x; dp[1] = (_Float16)v.y;
    dp[2] = (_Float16)v.z; dp[3] = (_Float16)v.w;
  }
  // stage W swizzled: element (k,n) -> sB[((k>>4)*64 + n)*16 + (k&15)]
  for (int i = tid; i < 64 * 16; i += 256) {
    int k = i >> 4, n4 = (i & 15) * 4;
    float4 v = *(const float4*)(W + k * 64 + n4);
    int base = (k >> 4) * 1024 + (k & 15);
    sB[base + (n4 + 0) * 16] = (_Float16)v.x;
    sB[base + (n4 + 1) * 16] = (_Float16)v.y;
    sB[base + (n4 + 2) * 16] = (_Float16)v.z;
    sB[base + (n4 + 3) * 16] = (_Float16)v.w;
  }
  __syncthreads();

  const int wave = tid >> 5;           // 0..7
  const int lane = tid & 31;
  const int n0   = (wave & 3) * 16;    // col tile
  const int mrow = lane & 15;          // M (A) / N (B,D) index
  const int lhi  = lane >> 4;          // lane group (0: lanes 0-15, 1: 16-31)
  const bool full = (m0 + 64 <= GN);   // block-uniform

  #pragma unroll
  for (int rr = 0; rr < 2; ++rr) {
    const int mbase = ((wave >> 2) + rr * 2) * 16;   // row tile base within block
    v8f acc = {};
    #pragma unroll
    for (int s = 0; s < 2; ++s) {      // K blocks of 32
      v16h a, b;
      // A fragment in this code's contiguous k order (same as B): lane<16 holds K {0..15}; lane>=16 holds {16..31}
      const int kbaseA = s * 32 + lhi * 16;
      const _Float16* arow = sA + (mbase + mrow) * 64;
      #pragma unroll
      for (int j = 0; j < 8; ++j)  a[j] = arow[kbaseA + j];          // contiguous 16B
      #pragma unroll
      for (int j = 8; j < 16; ++j) a[j] = arow[kbaseA + j];      // contiguous 16B
      // B fragment: 16 contiguous halves from swizzled LDS
      b = *(const v16h*)(sB + ((s * 2 + lhi) * 64 + n0 + mrow) * 16);
      acc = __builtin_amdgcn_wmma_f32_16x16x32_f16(
          /*neg_a=*/false, a, /*neg_b=*/false, b,
          /*c_mod=*/(short)0, acc, /*reuse_a=*/false, /*reuse_b=*/false);
    }
    // D layout: VGPR r -> row mbase + lhi*8 + r, col n0 + mrow
    const int row0 = m0 + mbase + lhi * 8;
    float* outp = Hout + (size_t)row0 * GH + n0 + mrow;
    if (full) {
      #pragma unroll
      for (int r = 0; r < 8; ++r) outp[r * GH] = acc[r];   // immediate offsets
    } else {
      #pragma unroll
      for (int r = 0; r < 8; ++r)
        if (row0 + r < GN) outp[r * GH] = acc[r];
    }
  }
}

// ---------------------------------------------------------------------------
// scatter: agg[dst] += h[src] * dinv[src]*dinv[dst]; 4 threads per edge,
// 16 channels each (float4 gathers, f32 atomics — L2-resident working set).
__global__ void scatter_kernel(const float* __restrict__ Hm, const int* __restrict__ src,
                               const int* __restrict__ dst, const float* __restrict__ dinv,
                               float* __restrict__ agg, int e_cnt) {
  unsigned t = blockIdx.x * blockDim.x + threadIdx.x;
  unsigned e = t >> 2;
  if (e >= (unsigned)e_cnt) return;
  int c0 = (t & 3) * 16;
  int s = src[e], d = dst[e];
  float ne = dinv[s] * dinv[d];
  const float4* hp = (const float4*)(Hm + (size_t)s * GH + c0);
  float* ap = agg + (size_t)d * GH + c0;
  #pragma unroll
  for (int q = 0; q < 4; ++q) {
    float4 v = hp[q];
    atomicAdd(ap + q * 4 + 0, v.x * ne);
    atomicAdd(ap + q * 4 + 1, v.y * ne);
    atomicAdd(ap + q * 4 + 2, v.z * ne);
    atomicAdd(ap + q * 4 + 3, v.w * ne);
  }
}

// ---------------------------------------------------------------------------
// out = agg + h*dinv^2 + bias, then optional BN(eval) + ReLU
__global__ void combine_kernel(const float* __restrict__ agg, const float* __restrict__ Hm,
                               const float* __restrict__ dinv, const float* __restrict__ bias,
                               const float* __restrict__ gamma, const float* __restrict__ beta,
                               const float* __restrict__ mean, const float* __restrict__ var,
                               float* __restrict__ out, int do_bn) {
  long long t = (long long)blockIdx.x * blockDim.x + threadIdx.x;
  if (t >= (long long)GN * GH) return;
  int n = (int)(t >> 6), c = (int)(t & 63);
  float ns = dinv[n]; ns *= ns;
  float v = agg[t] + Hm[t] * ns + bias[c];
  if (do_bn) {
    v = gamma[c] * (v - mean[c]) * rsqrtf(var[c] + GEPS) + beta[c];
    v = fmaxf(v, 0.0f);
  }
  out[t] = v;
}

// ---------------------------------------------------------------------------
// pooling: per-graph sums + counts via atomics
__global__ void pool_sum_kernel(const float* __restrict__ x, const int* __restrict__ batch,
                                float* __restrict__ sums) {
  long long t = (long long)blockIdx.x * blockDim.x + threadIdx.x;
  if (t >= (long long)GN * GH) return;
  int n = (int)(t >> 6), c = (int)(t & 63);
  atomicAdd(&sums[(size_t)batch[n] * GH + c], x[t]);
}

__global__ void pool_count_kernel(const int* __restrict__ batch, float* __restrict__ counts) {
  int n = blockIdx.x * blockDim.x + threadIdx.x;
  if (n < GN) atomicAdd(&counts[batch[n]], 1.0f);
}

// final tiny GEMM: out[g,o] = (sums[g,:]/max(cnt,1)) @ lin_W + lin_b
__global__ void pool_gemm_kernel(const float* __restrict__ sums, const float* __restrict__ counts,
                                 const float* __restrict__ lw, const float* __restrict__ lb,
                                 float* __restrict__ out) {
  int t = blockIdx.x * blockDim.x + threadIdx.x;   // G*OUT
  if (t >= GG * GOUT) return;
  int g = t >> 5, o = t & 31;
  float inv = 1.0f / fmaxf(counts[g], 1.0f);
  float acc = lb[o];
  #pragma unroll 8
  for (int k = 0; k < GH; ++k) acc += sums[g * GH + k] * inv * lw[k * GOUT + o];
  out[t] = acc;
}

// ---------------------------------------------------------------------------
extern "C" void kernel_launch(void* const* d_in, const int* in_sizes, int n_in,
                              void* d_out, int out_size, void* d_ws, size_t ws_size,
                              hipStream_t stream) {
  const float* x_in     = (const float*)d_in[0];
  const int*   eidx     = (const int*)d_in[1];
  const int*   batch    = (const int*)d_in[2];
  const float* Ws       = (const float*)d_in[3];
  const float* bs       = (const float*)d_in[4];
  const float* bn_gamma = (const float*)d_in[5];
  const float* bn_beta  = (const float*)d_in[6];
  const float* bn_mean  = (const float*)d_in[7];
  const float* bn_var   = (const float*)d_in[8];
  const float* lin_W    = (const float*)d_in[9];
  const float* lin_b    = (const float*)d_in[10];
  float* out = (float*)d_out;

  const int* src = eidx;
  const int* dst = eidx + GE;

  // workspace carve-up (floats):
  float* ws     = (float*)d_ws;
  float* dinv   = ws;                              // N
  float* hbuf   = dinv + GN;                       // N*H
  float* agg    = hbuf + (size_t)GN * GH;          // N*H
  float* xA     = agg  + (size_t)GN * GH;          // N*H
  float* xB     = xA   + (size_t)GN * GH;          // N*H
  float* sums   = xB   + (size_t)GN * GH;          // G*H
  float* counts = sums + (size_t)GG * GH;          // G

  const long long NH = (long long)GN * GH;
  const int TB = 256;

  // --- degree -> dinv ---
  zero_kernel<<<(GN + TB - 1) / TB, TB, 0, stream>>>(dinv, GN);
  degree_kernel<<<(GE + TB - 1) / TB, TB, 0, stream>>>(dst, dinv, GE);
  dinv_kernel<<<(GN + TB - 1) / TB, TB, 0, stream>>>(dinv, GN);

  // --- 3 GCN layers ---
  const float* xcur = x_in;
  float* xnext = xA;
  for (int layer = 0; layer < 3; ++layer) {
    gemm_xw_kernel<<<(GN + 63) / 64, 256, 0, stream>>>(xcur, Ws + (size_t)layer * GH * GH, hbuf);
    zero_kernel<<<(int)((NH + TB - 1) / TB), TB, 0, stream>>>(agg, NH);
    scatter_kernel<<<(GE * 4 + TB - 1) / TB, TB, 0, stream>>>(hbuf, src, dst, dinv, agg, GE);
    int bn_idx = (layer < 2) ? layer : 0;   // unused pointers for last layer stay in-bounds
    combine_kernel<<<(int)((NH + TB - 1) / TB), TB, 0, stream>>>(
        agg, hbuf, dinv, bs + (size_t)layer * GH,
        bn_gamma + (size_t)bn_idx * GH, bn_beta + (size_t)bn_idx * GH,
        bn_mean + (size_t)bn_idx * GH, bn_var + (size_t)bn_idx * GH,
        xnext, (layer < 2) ? 1 : 0);
    xcur = xnext;
    xnext = (xnext == xA) ? xB : xA;
  }

  // --- global mean pool + final linear ---
  zero_kernel<<<(GG * GH + GG + TB - 1) / TB, TB, 0, stream>>>(sums, GG * GH + GG); // sums+counts contiguous
  pool_sum_kernel<<<(int)((NH + TB - 1) / TB), TB, 0, stream>>>(xcur, batch, sums);
  pool_count_kernel<<<(GN + TB - 1) / TB, TB, 0, stream>>>(batch, counts);
  pool_gemm_kernel<<<(GG * GOUT + TB - 1) / TB, TB, 0, stream>>>(sums, counts, lin_W, lin_b, out);
}
